// TemporalAttentionFusion_14259291423254
// MI455X (gfx1250) — compile-verified, matrix operand order checked by reading, not run on hardware
//
#include <hip/hip_runtime.h>
#include <hip/hip_bf16.h>

typedef __bf16 bf16;
typedef __attribute__((ext_vector_type(16))) __bf16 v16bf;
typedef __attribute__((ext_vector_type(8)))  float  v8f;

union FragAB {
    v16bf v;
    uint4 q[2];
    bf16  e[16];
};
union Pack8 {
    bf16  e[8];
    uint4 q;
};

// Padded NHWC geometry: elem(b,y,x,c) at ((b*66 + y+1)*66 + (x+1))*C + c
#define PITCH 66

#define WMMA_BF16(A, B, C) \
    __builtin_amdgcn_wmma_f32_16x16x32_bf16(false, (A), false, (B), (short)0, (C), false, false)

// ---------------------------------------------------------------------------
// zero fill (padded buffers: borders stay zero, interiors overwritten)
// ---------------------------------------------------------------------------
__global__ void fill0_bf16(bf16* __restrict__ p, int n) {
    int i = blockIdx.x * blockDim.x + threadIdx.x;
    if (i < n) p[i] = (bf16)0.0f;
}

// ---------------------------------------------------------------------------
// x: NCHW f32 [8][C][64][64] -> padded NHWC bf16
// ---------------------------------------------------------------------------
__global__ void nchw_to_nhwcpad(const float* __restrict__ src,
                                bf16* __restrict__ dst, int C) {
    int idx = blockIdx.x * blockDim.x + threadIdx.x;   // (b*4096+hw)*C + c
    if (idx >= 8 * 4096 * C) return;
    int c  = idx % C;
    int n  = idx / C;
    int b  = n >> 12, hw = n & 4095;
    int h  = hw >> 6, w = hw & 63;
    dst[((size_t)(b * PITCH + h + 1) * PITCH + w + 1) * C + c] =
        (bf16)src[(((size_t)b * C + c) << 12) + hw];
}

// ---------------------------------------------------------------------------
// weights [O][C][3][3] f32 -> [r][O][C] bf16
// ---------------------------------------------------------------------------
__global__ void wperm9(const float* __restrict__ src, bf16* __restrict__ dst,
                       int O, int C) {
    int idx = blockIdx.x * blockDim.x + threadIdx.x;   // (o*C + c)*9 + r
    if (idx >= O * C * 9) return;
    int r  = idx % 9;
    int oc = idx / 9;
    int c  = oc % C;
    int o  = oc / C;
    dst[((size_t)r * O + o) * C + c] = (bf16)src[idx];
}

// w_dcn [64][64][3][3] f32 -> [m][g][k][c8] bf16 (row m: K index g*72+k*8+c)
__global__ void wperm_dcn(const float* __restrict__ src, bf16* __restrict__ dst) {
    int idx = blockIdx.x * blockDim.x + threadIdx.x;   // (m*64 + cin)*9 + k
    if (idx >= 64 * 64 * 9) return;
    int k   = idx % 9;
    int mc  = idx / 9;
    int cin = mc % 64;
    int m   = mc / 64;
    int g = cin >> 3, c = cin & 7;
    dst[(size_t)m * 576 + g * 72 + k * 8 + c] = (bf16)src[idx];
}

// ---------------------------------------------------------------------------
// Implicit-GEMM 3x3 conv, padded-NHWC bf16, tap-major weights.
// Register-blocked: each wave computes 16(M) x 64(N) (4 accumulators),
// A fragment reused across 4 B fragments per k-step.
// USE_LDS specialization: A fragments served from a per-block LDS weight
// slice ([9][32][ctot]); otherwise streamed from global. Channel loop is
// split at the concat boundary so there are no per-iteration selects.
// ---------------------------------------------------------------------------
extern __shared__ __align__(16) bf16 convlds[];

template <bool USE_LDS>
__device__ __forceinline__ void conv_accum(
    const bf16* __restrict__ in1, const bf16* __restrict__ in2,
    int cin1, int cin2, int ctot,
    const bf16* __restrict__ wgt, int cout, int m, int mloc, int half,
    const size_t* pix1, const size_t* pix2,
    v8f& acc0, v8f& acc1, v8f& acc2, v8f& acc3)
{
#pragma unroll
    for (int ry = 0; ry < 3; ++ry) {
#pragma unroll
        for (int rx = 0; rx < 3; ++rx) {
            const int r = ry * 3 + rx;
            const size_t toff = (size_t)ry * PITCH + rx;
            const bf16* wr = wgt + ((size_t)r * cout + m) * ctot;
            const size_t lbase = ((size_t)r * 32 + mloc) * ctot;

            const bf16* p1[4];
#pragma unroll
            for (int t = 0; t < 4; ++t) p1[t] = in1 + (pix1[t] + toff) * cin1;

            // channels [0, cin1) from in1
            for (int cb = 0; cb < cin1; cb += 32) {
                FragAB a, bm0, bm1, bm2, bm3;
                if (USE_LDS) {
                    a.q[0] = *(const uint4*)&convlds[lbase + cb + half * 8];
                    a.q[1] = *(const uint4*)&convlds[lbase + cb + 16 + half * 8];
                } else {
                    a.q[0] = *(const uint4*)(wr + cb + half * 8);
                    a.q[1] = *(const uint4*)(wr + cb + 16 + half * 8);
                }
                bm0.q[0] = *(const uint4*)(p1[0] + cb + half * 8);
                bm0.q[1] = *(const uint4*)(p1[0] + cb + 16 + half * 8);
                bm1.q[0] = *(const uint4*)(p1[1] + cb + half * 8);
                bm1.q[1] = *(const uint4*)(p1[1] + cb + 16 + half * 8);
                bm2.q[0] = *(const uint4*)(p1[2] + cb + half * 8);
                bm2.q[1] = *(const uint4*)(p1[2] + cb + 16 + half * 8);
                bm3.q[0] = *(const uint4*)(p1[3] + cb + half * 8);
                bm3.q[1] = *(const uint4*)(p1[3] + cb + 16 + half * 8);
                acc0 = WMMA_BF16(a.v, bm0.v, acc0);
                acc1 = WMMA_BF16(a.v, bm1.v, acc1);
                acc2 = WMMA_BF16(a.v, bm2.v, acc2);
                acc3 = WMMA_BF16(a.v, bm3.v, acc3);
            }

            if (cin2 > 0) {
                const bf16* p2[4];
#pragma unroll
                for (int t = 0; t < 4; ++t) p2[t] = in2 + (pix2[t] + toff) * cin2;
                // channels [cin1, ctot) from in2
                for (int cb = 0; cb < cin2; cb += 32) {
                    const int ck = cin1 + cb;   // global K offset
                    FragAB a, bm0, bm1, bm2, bm3;
                    if (USE_LDS) {
                        a.q[0] = *(const uint4*)&convlds[lbase + ck + half * 8];
                        a.q[1] = *(const uint4*)&convlds[lbase + ck + 16 + half * 8];
                    } else {
                        a.q[0] = *(const uint4*)(wr + ck + half * 8);
                        a.q[1] = *(const uint4*)(wr + ck + 16 + half * 8);
                    }
                    bm0.q[0] = *(const uint4*)(p2[0] + cb + half * 8);
                    bm0.q[1] = *(const uint4*)(p2[0] + cb + 16 + half * 8);
                    bm1.q[0] = *(const uint4*)(p2[1] + cb + half * 8);
                    bm1.q[1] = *(const uint4*)(p2[1] + cb + 16 + half * 8);
                    bm2.q[0] = *(const uint4*)(p2[2] + cb + half * 8);
                    bm2.q[1] = *(const uint4*)(p2[2] + cb + 16 + half * 8);
                    bm3.q[0] = *(const uint4*)(p2[3] + cb + half * 8);
                    bm3.q[1] = *(const uint4*)(p2[3] + cb + 16 + half * 8);
                    acc0 = WMMA_BF16(a.v, bm0.v, acc0);
                    acc1 = WMMA_BF16(a.v, bm1.v, acc1);
                    acc2 = WMMA_BF16(a.v, bm2.v, acc2);
                    acc3 = WMMA_BF16(a.v, bm3.v, acc3);
                }
            }
        }
    }
}

__global__ __launch_bounds__(256)
void conv3_wmma(const bf16* __restrict__ in1, const bf16* __restrict__ in2,
                int in2_fixed_b, int cin1, int cin2,
                const bf16* __restrict__ wgt,   // [9][cout][ctot]
                const float* __restrict__ bias,
                int cout, int act, int use_lds,
                bf16*  __restrict__ out_bf,     // padded NHWC [8][66][66][cout]
                float* __restrict__ out_nhwc,   // [32768][cout]
                float* __restrict__ out_nchw)   // [8][cout][4096]
{
    const int ctot = cin1 + cin2;
    const int lane = threadIdx.x & 31;
    const int wave = threadIdx.x >> 5;
    const int half = lane >> 4;
    const int l16  = lane & 15;
    const int tileM  = blockIdx.y * 32 + (wave >> 2) * 16;
    const int tileN0 = blockIdx.x * 256 + (wave & 3) * 64;
    const int mBase  = blockIdx.y * 32;

    int m = tileM + l16;
    if (m >= cout) m = cout - 1;

    if (use_lds) {
        // stage [9][32][ctot] weight slice (rows mBase..mBase+31, clamped)
        const int cpr   = ctot >> 3;            // 16B chunks per row
        const int total = 9 * 32 * cpr;
        for (int ch = threadIdx.x; ch < total; ch += 256) {
            int r   = ch / (32 * cpr);
            int rem = ch - r * 32 * cpr;
            int j   = rem / cpr;
            int cp  = rem - j * cpr;
            int row = mBase + j;
            if (row >= cout) row = cout - 1;
            *(uint4*)&convlds[((size_t)r * 32 + j) * ctot + cp * 8] =
                *(const uint4*)(wgt + ((size_t)r * cout + row) * ctot + cp * 8);
        }
        __syncthreads();
    }
    const int mloc = (tileM - mBase) + l16;      // 0..31 row within LDS slice

    int    nn[4], bb[4], hh[4], ww[4];
    size_t pix1[4], pix2[4];
#pragma unroll
    for (int t = 0; t < 4; ++t) {
        int n = tileN0 + t * 16 + l16;
        nn[t] = n;
        int b = n >> 12, hw = n & 4095;
        int h = hw >> 6, w = hw & 63;
        bb[t] = b; hh[t] = h; ww[t] = w;
        int b2 = (in2_fixed_b >= 0) ? in2_fixed_b : b;
        pix1[t] = (size_t)(b * PITCH + h) * PITCH + w;
        pix2[t] = (size_t)(b2 * PITCH + h) * PITCH + w;
    }

    v8f acc0 = {}, acc1 = {}, acc2 = {}, acc3 = {};
    if (use_lds) {
        conv_accum<true >(in1, in2, cin1, cin2, ctot, wgt, cout, m, mloc, half,
                          pix1, pix2, acc0, acc1, acc2, acc3);
    } else {
        conv_accum<false>(in1, in2, cin1, cin2, ctot, wgt, cout, m, mloc, half,
                          pix1, pix2, acc0, acc1, acc2, acc3);
    }

    v8f accs[4] = {acc0, acc1, acc2, acc3};
#pragma unroll
    for (int t = 0; t < 4; ++t) {
        float vals[8];
#pragma unroll
        for (int r = 0; r < 8; ++r) {
            int mm = tileM + half * 8 + r;
            float vv = accs[t][r] + bias[mm < cout ? mm : cout - 1];
            if (act) vv = fmaxf(vv, 0.f);
            vals[r] = vv;
        }
        if (out_bf) {   // cout is a multiple of 32 on this path
            Pack8 pk;
#pragma unroll
            for (int r = 0; r < 8; ++r) pk.e[r] = (bf16)vals[r];
            *(uint4*)(out_bf
                + ((size_t)(bb[t] * PITCH + hh[t] + 1) * PITCH + ww[t] + 1) * cout
                + tileM + half * 8) = pk.q;
        }
        if (out_nhwc) {
#pragma unroll
            for (int r = 0; r < 8; ++r) {
                int mm = tileM + half * 8 + r;
                if (mm < cout) out_nhwc[(size_t)nn[t] * cout + mm] = vals[r];
            }
        }
        if (out_nchw) {
#pragma unroll
            for (int r = 0; r < 8; ++r) {
                int mm = tileM + half * 8 + r;
                if (mm < cout)
                    out_nchw[(((size_t)bb[t] * cout + mm) << 12) + (hh[t] << 6) + ww[t]] = vals[r];
            }
        }
    }
}

// ---------------------------------------------------------------------------
// DCN sampler: S[n][g*72 + k*8 + c] = sigmoid(mask)*bilinear(f[b, :, :, g*8+c])
// ---------------------------------------------------------------------------
__global__ void dcn_build(const float* __restrict__ off,
                          const bf16*  __restrict__ f,
                          bf16* __restrict__ S)
{
    int idx = blockIdx.x * blockDim.x + threadIdx.x;    // n*72 + g9k
    if (idx >= 32768 * 72) return;
    int g9k = idx % 72;
    int n   = idx / 72;
    int g = g9k / 9;
    int k = g9k - g * 9;
    int b = n >> 12, hw = n & 4095;
    int h = hw >> 6, w = hw & 63;

    const float* orow = off + (size_t)n * 216;
    float dy = orow[g9k];
    float dx = orow[72 + g9k];
    float mk = orow[144 + g9k];
    float mval = 1.f / (1.f + __expf(-mk));

    float py = dy + (float)(h - 1 + k / 3);
    float px = dx + (float)(w - 1 + (k - (k / 3) * 3));
    float y0f = floorf(py), x0f = floorf(px);
    int y0 = (int)y0f, x0 = (int)x0f;
    int y1 = y0 + 1,   x1 = x0 + 1;
    float ty = py - y0f, tx = px - x0f;

    float w00 = (1.f - ty) * (1.f - tx), w01 = (1.f - ty) * tx;
    float w10 = ty * (1.f - tx),         w11 = ty * tx;
    bool vy0 = (y0 >= 0 && y0 <= 63), vy1 = (y1 >= 0 && y1 <= 63);
    bool vx0 = (x0 >= 0 && x0 <= 63), vx1 = (x1 >= 0 && x1 <= 63);
    int y0c = min(max(y0, 0), 63), y1c = min(max(y1, 0), 63);
    int x0c = min(max(x0, 0), 63), x1c = min(max(x1, 0), 63);
    if (!(vy0 && vx0)) w00 = 0.f;
    if (!(vy0 && vx1)) w01 = 0.f;
    if (!(vy1 && vx0)) w10 = 0.f;
    if (!(vy1 && vx1)) w11 = 0.f;

    const bf16* fb = f + g * 8;
    auto pix = [&](int y, int x) {
        return fb + ((size_t)(b * PITCH + y + 1) * PITCH + x + 1) * 64;
    };
    Pack8 c00, c01, c10, c11, so;
    c00.q = *(const uint4*)pix(y0c, x0c);
    c01.q = *(const uint4*)pix(y0c, x1c);
    c10.q = *(const uint4*)pix(y1c, x0c);
    c11.q = *(const uint4*)pix(y1c, x1c);
#pragma unroll
    for (int c = 0; c < 8; ++c) {
        float v = w00 * (float)c00.e[c] + w01 * (float)c01.e[c]
                + w10 * (float)c10.e[c] + w11 * (float)c11.e[c];
        so.e[c] = (bf16)(v * mval);
    }
    *(uint4*)(S + (size_t)n * 576 + g * 72 + k * 8) = so.q;
}

// ---------------------------------------------------------------------------
// DCN einsum GEMM: out = (W[64x576] x S[576x32768] + bias) * f[iref]
// Whole W staged in LDS via CDNA5 async-to-LDS (ASYNCcnt), A from ds_load.
// ---------------------------------------------------------------------------
__global__ __launch_bounds__(256)
void dcn_gemm_wmma(const bf16* __restrict__ S,     // [32768][576]
                   const bf16* __restrict__ wgt,   // [64][576]
                   const float* __restrict__ bias,
                   const bf16* __restrict__ f,     // padded NHWC [8][66][66][64]
                   int iref,
                   bf16* __restrict__ out)         // padded NHWC [8][66][66][64]
{
    __shared__ __align__(16) bf16 wlds[64 * 576];  // 73728 B

    // Async stage of the full weight matrix: 4608 x 16B chunks.
    {
        unsigned lbase = (unsigned)(uintptr_t)(&wlds[0]);
        unsigned long long gbase = (unsigned long long)(uintptr_t)wgt;
#pragma unroll
        for (int t = 0; t < 18; ++t) {
            unsigned chunk = threadIdx.x + t * 256;
            unsigned laddr = lbase + chunk * 16;
            unsigned long long gaddr = gbase + (unsigned long long)chunk * 16;
            asm volatile("global_load_async_to_lds_b128 %0, %1, off"
                         :: "v"(laddr), "v"(gaddr) : "memory");
        }
        asm volatile("s_wait_asynccnt 0" ::: "memory");
        __syncthreads();
    }

    const int lane = threadIdx.x & 31;
    const int wave = threadIdx.x >> 5;
    const int half = lane >> 4;
    const int l16  = lane & 15;
    const int tileM  = blockIdx.y * 32 + (wave >> 2) * 16;
    const int tileN0 = blockIdx.x * 256 + (wave & 3) * 64;
    const int m = tileM + l16;
    const bf16* srow[4];
    int nn[4];
#pragma unroll
    for (int t = 0; t < 4; ++t) {
        nn[t] = tileN0 + t * 16 + l16;
        srow[t] = S + (size_t)nn[t] * 576;
    }

    v8f acc0 = {}, acc1 = {}, acc2 = {}, acc3 = {};
#pragma unroll 2
    for (int kbase = 0; kbase < 576; kbase += 32) {
        FragAB a, bm0, bm1, bm2, bm3;
        a.q[0]   = *(const uint4*)&wlds[(size_t)m * 576 + kbase + half * 8];
        a.q[1]   = *(const uint4*)&wlds[(size_t)m * 576 + kbase + 16 + half * 8];
        bm0.q[0] = *(const uint4*)(srow[0] + kbase + half * 8);
        bm0.q[1] = *(const uint4*)(srow[0] + kbase + 16 + half * 8);
        bm1.q[0] = *(const uint4*)(srow[1] + kbase + half * 8);
        bm1.q[1] = *(const uint4*)(srow[1] + kbase + 16 + half * 8);
        bm2.q[0] = *(const uint4*)(srow[2] + kbase + half * 8);
        bm2.q[1] = *(const uint4*)(srow[2] + kbase + 16 + half * 8);
        bm3.q[0] = *(const uint4*)(srow[3] + kbase + half * 8);
        bm3.q[1] = *(const uint4*)(srow[3] + kbase + 16 + half * 8);
        acc0 = WMMA_BF16(a.v, bm0.v, acc0);
        acc1 = WMMA_BF16(a.v, bm1.v, acc1);
        acc2 = WMMA_BF16(a.v, bm2.v, acc2);
        acc3 = WMMA_BF16(a.v, bm3.v, acc3);
    }

    v8f accs[4] = {acc0, acc1, acc2, acc3};
#pragma unroll
    for (int t = 0; t < 4; ++t) {
        const int b = nn[t] >> 12, hw = nn[t] & 4095;
        const int h = hw >> 6, w = hw & 63;
        Pack8 mr, pk;
        mr.q = *(const uint4*)(f + ((size_t)(iref * PITCH + h + 1) * PITCH + w + 1) * 64
                                 + tileM + half * 8);
#pragma unroll
        for (int r = 0; r < 8; ++r) {
            float vv = (accs[t][r] + bias[tileM + half * 8 + r]) * (float)mr.e[r];
            pk.e[r] = (bf16)vv;
        }
        *(uint4*)(out + ((size_t)(b * PITCH + h + 1) * PITCH + w + 1) * 64
                      + tileM + half * 8) = pk.q;
    }
}

// ---------------------------------------------------------------------------
// Softmax over T=8 frames + weighted accumulation with f -> feats[iref]
// ---------------------------------------------------------------------------
__global__ void softmax_fuse(const float* __restrict__ xc, // NHWC [8][4096][64]
                             const bf16*  __restrict__ f,  // padded NHWC
                             bf16* __restrict__ ostack,    // padded NHWC
                             int iref)
{
    int idx = blockIdx.x * blockDim.x + threadIdx.x;       // hw*64 + c
    if (idx >= (4096 * 64)) return;
    int c  = idx & 63;
    int hw = idx >> 6;
    int h  = hw >> 6, w = hw & 63;

    float vmax = -3.4e38f;
#pragma unroll
    for (int t = 0; t < 8; ++t)
        vmax = fmaxf(vmax, xc[((size_t)t << 18) + idx]);
    float ssum = 0.f, accw = 0.f;
#pragma unroll
    for (int t = 0; t < 8; ++t) {
        float e = __expf(xc[((size_t)t << 18) + idx] - vmax);
        ssum += e;
        accw += e * (float)f[((size_t)(t * PITCH + h + 1) * PITCH + w + 1) * 64 + c];
    }
    ostack[((size_t)(iref * PITCH + h + 1) * PITCH + w + 1) * 64 + c] =
        (bf16)(accw / ssum);
}

// ---------------------------------------------------------------------------
// Host sequencing
// ---------------------------------------------------------------------------
extern "C" void kernel_launch(void* const* d_in, const int* in_sizes, int n_in,
                              void* d_out, int out_size, void* d_ws, size_t ws_size,
                              hipStream_t stream) {
    const float* x     = (const float*)d_in[0];
    const float* w1    = (const float*)d_in[1];
    const float* b1    = (const float*)d_in[2];
    const float* w_off = (const float*)d_in[3];
    const float* b_off = (const float*)d_in[4];
    const float* w_co  = (const float*)d_in[5];
    const float* b_co  = (const float*)d_in[6];
    const float* w_dcn = (const float*)d_in[7];
    const float* b_dcn = (const float*)d_in[8];
    const float* w_emb = (const float*)d_in[9];
    const float* b_emb = (const float*)d_in[10];
    const float* w2    = (const float*)d_in[11];
    const float* b2    = (const float*)d_in[12];
    float* out = (float*)d_out;

    char* ws = (char*)d_ws;
    auto carve = [&](size_t bytes) {
        void* p = (void*)ws;
        ws += (bytes + 255) & ~(size_t)255;
        return p;
    };
    const int PADPIX = 8 * PITCH * PITCH;                 // 34848 padded pixels
    bf16* xbf     = (bf16*)carve((size_t)PADPIX * 256 * 2);
    bf16* w1p     = (bf16*)carve((size_t)9 * 64 * 256 * 2);
    bf16* w_offp  = (bf16*)carve((size_t)9 * 64 * 128 * 2);
    bf16* w_cop   = (bf16*)carve((size_t)9 * 216 * 64 * 2);
    bf16* w_dcnp  = (bf16*)carve((size_t)64 * 576 * 2);
    bf16* w_embp  = (bf16*)carve((size_t)3 * 9 * 64 * 64 * 2);
    bf16* w2p     = (bf16*)carve((size_t)9 * 256 * 64 * 2);
    bf16* f_bf    = (bf16*)carve((size_t)PADPIX * 64 * 2);
    bf16* xset_bf = (bf16*)carve((size_t)PADPIX * 64 * 2);
    float* off_nh = (float*)carve((size_t)32768 * 216 * 4);
    bf16* S       = (bf16*)carve((size_t)32768 * 576 * 2);
    bf16* xcor_a  = (bf16*)carve((size_t)PADPIX * 64 * 2);
    bf16* xcor_b  = (bf16*)carve((size_t)PADPIX * 64 * 2);
    float* xcor_f = (float*)carve((size_t)32768 * 64 * 4);
    bf16* ostack  = (bf16*)carve((size_t)PADPIX * 64 * 2);

    const dim3 blk(256);
    auto fill = [&](bf16* p, int n) {
        fill0_bf16<<<(n + 255) / 256, blk, 0, stream>>>(p, n);
    };
    fill(xbf, PADPIX * 256);
    fill(f_bf, PADPIX * 64);
    fill(xset_bf, PADPIX * 64);
    fill(xcor_a, PADPIX * 64);
    fill(xcor_b, PADPIX * 64);
    fill(ostack, PADPIX * 64);

    nchw_to_nhwcpad<<<(8 * 4096 * 256 + 255) / 256, blk, 0, stream>>>(x, xbf, 256);
    wperm9<<<(64 * 256 * 9 + 255) / 256, blk, 0, stream>>>(w1, w1p, 64, 256);
    wperm9<<<(64 * 128 * 9 + 255) / 256, blk, 0, stream>>>(w_off, w_offp, 64, 128);
    wperm9<<<(216 * 64 * 9 + 255) / 256, blk, 0, stream>>>(w_co, w_cop, 216, 64);
    for (int j = 0; j < 3; ++j)
        wperm9<<<(64 * 64 * 9 + 255) / 256, blk, 0, stream>>>(
            w_emb + (size_t)j * 64 * 64 * 9, w_embp + (size_t)j * 9 * 64 * 64, 64, 64);
    wperm9<<<(256 * 64 * 9 + 255) / 256, blk, 0, stream>>>(w2, w2p, 256, 64);
    wperm_dcn<<<(64 * 64 * 9 + 255) / 256, blk, 0, stream>>>(w_dcn, w_dcnp);

    const int gx = 128;   // 32768 pixels / 256 per block
    const size_t lds64  = (size_t)9 * 32 * 64 * 2;    // 36864 B
    const size_t lds128 = (size_t)9 * 32 * 128 * 2;   // 73728 B

    // f = relu(conv1(x))  (ctot=256: streamed-global weights, no LDS)
    conv3_wmma<<<dim3(gx, 2), blk, 0, stream>>>(
        xbf, nullptr, -1, 256, 0, w1p, b1, 64, 1, 0, f_bf, nullptr, nullptr);

    for (int i = 0; i < 8; ++i) {
        // x_set = conv3(concat(f, f[i]))
        conv3_wmma<<<dim3(gx, 2), blk, lds128, stream>>>(
            f_bf, f_bf, i, 64, 64, w_offp, b_off, 64, 0, 1, xset_bf, nullptr, nullptr);
        // off_out (216 ch, f32 NHWC)
        conv3_wmma<<<dim3(gx, 7), blk, lds64, stream>>>(
            xset_bf, nullptr, -1, 64, 0, w_cop, b_co, 216, 0, 1,
            nullptr, off_nh, nullptr);
        // S = modulated bilinear samples of f
        dcn_build<<<(32768 * 72 + 255) / 256, blk, 0, stream>>>(off_nh, f_bf, S);
        // x_cor = (W_dcn x S + b) * f[i]
        dcn_gemm_wmma<<<dim3(gx, 2), blk, 0, stream>>>(
            S, w_dcnp, b_dcn, f_bf, i, xcor_a);
        // 3 emb convs, last to f32 for softmax
        conv3_wmma<<<dim3(gx, 2), blk, lds64, stream>>>(
            xcor_a, nullptr, -1, 64, 0, w_embp, b_emb, 64, 0, 1,
            xcor_b, nullptr, nullptr);
        conv3_wmma<<<dim3(gx, 2), blk, lds64, stream>>>(
            xcor_b, nullptr, -1, 64, 0, w_embp + 9 * 64 * 64, b_emb + 64, 64, 0, 1,
            xcor_a, nullptr, nullptr);
        conv3_wmma<<<dim3(gx, 2), blk, lds64, stream>>>(
            xcor_a, nullptr, -1, 64, 0, w_embp + 2 * 9 * 64 * 64, b_emb + 128, 64, 0, 1,
            nullptr, xcor_f, nullptr);
        // softmax over frames + weighted sum -> feats[i]
        softmax_fuse<<<(4096 * 64 + 255) / 256, blk, 0, stream>>>(
            xcor_f, f_bf, ostack, i);
    }

    // out = relu(conv3(stack, w2)) -> NCHW f32 d_out
    conv3_wmma<<<dim3(gx, 8), blk, lds64, stream>>>(
        ostack, nullptr, -1, 64, 0, w2p, b2, 256, 1, 1, nullptr, nullptr, out);
}
